// RGCN_5153960755640
// MI455X (gfx1250) — matrix operand order repaired and checked by reading, not compiled, not run on hardware
//
#include <hip/hip_runtime.h>

// ---------------- constants from the reference problem ----------------
#define H128 128   // hidden width (layers 0/1 output, layer-2 input)
#define RREL 8     // num relations
#define BBAS 4     // num bases
#define CCLS 16    // num classes

typedef __attribute__((ext_vector_type(16))) __bf16 bf16x16;
typedef __attribute__((ext_vector_type(8)))  __bf16 bf16x8;
typedef __attribute__((ext_vector_type(8)))  float  f32x8;

// ---------------- CDNA5 packed bf16 global atomic (scatter into z) ----
__device__ __forceinline__ void atomic_pk_add_bf16(__bf16* addr, unsigned pk) {
  asm volatile("global_atomic_pk_add_bf16 %0, %1, off" : : "v"(addr), "v"(pk) : "memory");
}

// A-fragment (16x32 bf16, ISA layout: lanes0-15 rows, K interleave 0..7 / 16..23,
// upper half-wave offset +8 in K)
__device__ __forceinline__ bf16x16 load_fragA(const __bf16* p) {
  bf16x8 lo = *(const bf16x8*)(p);        // K = koff .. koff+7
  bf16x8 hi = *(const bf16x8*)(p + 16);   // K = koff+16 .. koff+23
  return __builtin_shufflevector(lo, hi, 0,1,2,3,4,5,6,7,8,9,10,11,12,13,14,15);
}

// ---------------- zero scratch (deterministic per call) ---------------
__global__ void zero16_kernel(uint4* __restrict__ p, size_t n16) {
  size_t i = (size_t)blockIdx.x * blockDim.x + threadIdx.x;
  if (i < n16) p[i] = make_uint4(0u, 0u, 0u, 0u);
}

// ---------------- compose W_r^T = (sum_b wcomp[r,b]*bases[b])^T, bf16 --
// bases: [B, K, Nout] f32 ; out wT: [R, Nout, K] bf16 (row = output col, for WMMA B)
__global__ void compose_wT_kernel(const float* __restrict__ bases,
                                  const float* __restrict__ wcomp,
                                  __bf16* __restrict__ wT,
                                  int K, int Nout, size_t total) {
  size_t i = (size_t)blockIdx.x * blockDim.x + threadIdx.x;
  if (i >= total) return;
  int k = (int)(i % K);
  size_t t = i / K;
  int n = (int)(t % Nout);
  int r = (int)(t / Nout);
  float a = 0.f;
#pragma unroll
  for (int b = 0; b < BBAS; ++b)
    a += wcomp[r * BBAS + b] * bases[((size_t)b * K + k) * Nout + n];
  wT[i] = (__bf16)a;
}

// ---------------- transpose f32 [K,Nout] -> bf16 [Nout,K] --------------
__global__ void transpose_bf_kernel(const float* __restrict__ L,
                                    __bf16* __restrict__ T,
                                    int K, int Nout, size_t total) {
  size_t i = (size_t)blockIdx.x * blockDim.x + threadIdx.x;
  if (i >= total) return;
  int k = (int)(i % K);
  int n = (int)(i / K);
  T[i] = (__bf16)L[(size_t)k * Nout + n];
}

// ---------------- layer-0 edges: agg[dst] += norm * sum_b wc[r,b]*bases0[b,id[src]]
__global__ void l0_edge_kernel(const int* __restrict__ src, const int* __restrict__ dst,
                               const int* __restrict__ ids, const int* __restrict__ rel,
                               const float* __restrict__ norm,
                               const float* __restrict__ bases0,
                               const float* __restrict__ wcomp0,
                               float* __restrict__ agg, int E, int N) {
  int e = blockIdx.x * blockDim.y + threadIdx.y;   // one wave per edge
  if (e >= E) return;
  int lane = threadIdx.x;
  int s = ids[src[e]];
  int d = dst[e];
  int r = rel[e];
  float nrm = norm[e];
  float wc0 = wcomp0[r * BBAS + 0], wc1 = wcomp0[r * BBAS + 1];
  float wc2 = wcomp0[r * BBAS + 2], wc3 = wcomp0[r * BBAS + 3];
  const float4* b0 = (const float4*)(bases0 + ((size_t)0 * N + s) * H128);
  const float4* b1 = (const float4*)(bases0 + ((size_t)1 * N + s) * H128);
  const float4* b2 = (const float4*)(bases0 + ((size_t)2 * N + s) * H128);
  const float4* b3 = (const float4*)(bases0 + ((size_t)3 * N + s) * H128);
  float4 v0 = b0[lane], v1 = b1[lane], v2 = b2[lane], v3 = b3[lane];
  float4 m;
  m.x = nrm * (wc0 * v0.x + wc1 * v1.x + wc2 * v2.x + wc3 * v3.x);
  m.y = nrm * (wc0 * v0.y + wc1 * v1.y + wc2 * v2.y + wc3 * v3.y);
  m.z = nrm * (wc0 * v0.z + wc1 * v1.z + wc2 * v2.z + wc3 * v3.z);
  m.w = nrm * (wc0 * v0.w + wc1 * v1.w + wc2 * v2.w + wc3 * v3.w);
  float* out = agg + (size_t)d * H128 + lane * 4;
  atomicAdd(out + 0, m.x);
  atomicAdd(out + 1, m.y);
  atomicAdd(out + 2, m.z);
  atomicAdd(out + 3, m.w);
}

// ---------------- aggregate-first scatter: z[r][dst] += norm * x[src] (bf16 pk atomics)
__global__ void z_edge_kernel(const int* __restrict__ src, const int* __restrict__ dst,
                              const int* __restrict__ rel, const float* __restrict__ norm,
                              const float* __restrict__ x, __bf16* __restrict__ z,
                              int E, int N) {
  int e = blockIdx.x * blockDim.y + threadIdx.y;   // one wave per edge
  if (e >= E) return;
  int lane = threadIdx.x;
  int s = src[e];
  int d = dst[e];
  int r = rel[e];
  float nrm = norm[e];
  float4 v = ((const float4*)(x + (size_t)s * H128))[lane];
  __bf16* zp = z + ((size_t)r * N + d) * H128 + lane * 4;
  union { __bf16 h[2]; unsigned u; } p0, p1;
  p0.h[0] = (__bf16)(v.x * nrm); p0.h[1] = (__bf16)(v.y * nrm);
  p1.h[0] = (__bf16)(v.z * nrm); p1.h[1] = (__bf16)(v.w * nrm);
  atomic_pk_add_bf16(zp + 0, p0.u);
  atomic_pk_add_bf16(zp + 2, p1.u);
}

// ---------------- WMMA GEMM: OUT[m,n] = [Cin] + bias + sum_t A_t[m,:] . BT_t[n,:]
// A_t : bf16 [M,128] row-major at A + t*aStride
// BT_t: bf16 [Nout,128] row-major (i.e. B transposed) at BT + t*bStride
// Block = (32, Nout/16): one wave per 16x16 output tile; M must be multiple of 16.
__global__ void wmma_gemm_bf16(const __bf16* __restrict__ A, size_t aStride, int nTerms,
                               const __bf16* __restrict__ BT, size_t bStride,
                               const float* __restrict__ Cin,
                               const float* __restrict__ bias,
                               float* __restrict__ OUT, int Nout) {
  const int lane = threadIdx.x;
  const int mt = blockIdx.x << 4;
  const int half = lane >> 4;          // 0: lanes 0-15, 1: lanes 16-31
  const int l15 = lane & 15;
  const int koffA  = half << 4;        // A K offset: 0 / 16
  const int kbaseB = half << 4;        // B K base:   0 / 16
  const int nc = (threadIdx.y << 4) + l15;   // output column
  f32x8 acc;
#pragma unroll
  for (int i = 0; i < 8; ++i) acc[i] = 0.f;
  if (Cin) {
#pragma unroll
    for (int i = 0; i < 8; ++i)
      acc[i] = Cin[(size_t)(mt + (half << 3) + i) * Nout + nc];
  }
  const int rowA = mt + l15;
  for (int t = 0; t < nTerms; ++t) {
    const __bf16* Ar = A + (size_t)t * aStride + (size_t)rowA * H128 + koffA;
    const __bf16* Br = BT + (size_t)t * bStride + (size_t)nc * H128 + kbaseB;
#pragma unroll
    for (int kk = 0; kk < H128; kk += 32) {
      bf16x16 a = *(const bf16x16*)(Ar + kk);
      bf16x16 b = *(const bf16x16*)(Br + kk);
      acc = __builtin_amdgcn_wmma_f32_16x16x32_bf16(false, a, false, b,
                                                    (short)0, acc, false, false);
    }
  }
  float bv = bias ? bias[nc] : 0.f;
#pragma unroll
  for (int i = 0; i < 8; ++i)
    OUT[(size_t)(mt + (half << 3) + i) * Nout + nc] = acc[i] + bv;
}

// ---------------- epilogue: x = relu(acc [+ loop[ids]] [+ bias]) + bf16 copy
__global__ void finish_kernel(const float* __restrict__ acc, const float* __restrict__ loop,
                              const int* __restrict__ ids, const float* __restrict__ bias,
                              float* __restrict__ xout, __bf16* __restrict__ xbf,
                              size_t total) {
  size_t i = (size_t)blockIdx.x * blockDim.x + threadIdx.x;
  if (i >= total) return;
  int n = (int)(i >> 7);     // H == 128
  int f = (int)(i & 127);
  float v = acc[i];
  if (loop) {
    int row = ids ? ids[n] : n;
    v += loop[(size_t)row * H128 + f];
  }
  if (bias) v += bias[f];
  v = fmaxf(v, 0.f);
  xout[i] = v;
  xbf[i] = (__bf16)v;
}

// ======================================================================
extern "C" void kernel_launch(void* const* d_in, const int* in_sizes, int n_in,
                              void* d_out, int out_size, void* d_ws, size_t ws_size,
                              hipStream_t stream) {
  const int*   src    = (const int*)d_in[0];
  const int*   dst    = (const int*)d_in[1];
  const int*   h      = (const int*)d_in[2];
  const int*   rel    = (const int*)d_in[3];
  const float* norm   = (const float*)d_in[4];
  const float* bases0 = (const float*)d_in[5];
  const float* wcomp0 = (const float*)d_in[6];
  const float* loop0  = (const float*)d_in[7];
  const float* bias0  = (const float*)d_in[8];
  const float* bases1 = (const float*)d_in[9];
  const float* wcomp1 = (const float*)d_in[10];
  const float* loop1  = (const float*)d_in[11];
  const float* bias1  = (const float*)d_in[12];
  const float* bases2 = (const float*)d_in[13];
  const float* wcomp2 = (const float*)d_in[14];
  const float* loop2  = (const float*)d_in[15];
  const float* bias2  = (const float*)d_in[16];

  const int    E  = in_sizes[0];
  const int    N  = in_sizes[2];
  const size_t NH = (size_t)N * H128;

  // ---- workspace carve-up (~167 MB) ----
  char*  ws  = (char*)d_ws;
  size_t off = 0;
  auto take = [&](size_t bytes) -> char* {
    char* p = ws + off;
    off = (off + bytes + 255) & ~(size_t)255;
    return p;
  };
  float*  bufA = (float*)take(NH * sizeof(float));                    // agg0 / out1
  float*  bufB = (float*)take(NH * sizeof(float));                    // x1 / x2 (f32)
  __bf16* xbf  = (__bf16*)take(NH * 2);                               // x in bf16
  __bf16* zbf  = (__bf16*)take((size_t)RREL * NH * 2);                // z[r][node][H]
  __bf16* w1T  = (__bf16*)take((size_t)RREL * H128 * H128 * 2);
  __bf16* w2T  = (__bf16*)take((size_t)RREL * CCLS * H128 * 2);
  __bf16* l1T  = (__bf16*)take((size_t)H128 * H128 * 2);
  __bf16* l2T  = (__bf16*)take((size_t)CCLS * H128 * 2);
  (void)ws_size; (void)n_in;

  const dim3 edgeBlk(32, 8);
  const unsigned eBlocks = (unsigned)((E + 7) / 8);
  const unsigned nhBlocks = (unsigned)((NH + 255) / 256);
  const unsigned mBlocks = (unsigned)((N + 15) / 16);   // N % 16 == 0 for this problem

  // ---- compose relation weights (transposed, bf16) ----
  {
    size_t t1 = (size_t)RREL * H128 * H128;
    compose_wT_kernel<<<(unsigned)((t1 + 255) / 256), 256, 0, stream>>>(bases1, wcomp1, w1T, H128, H128, t1);
    size_t t2 = (size_t)RREL * CCLS * H128;
    compose_wT_kernel<<<(unsigned)((t2 + 255) / 256), 256, 0, stream>>>(bases2, wcomp2, w2T, H128, CCLS, t2);
    size_t tl1 = (size_t)H128 * H128;
    transpose_bf_kernel<<<(unsigned)((tl1 + 255) / 256), 256, 0, stream>>>(loop1, l1T, H128, H128, tl1);
    size_t tl2 = (size_t)CCLS * H128;
    transpose_bf_kernel<<<(unsigned)((tl2 + 255) / 256), 256, 0, stream>>>(loop2, l2T, H128, CCLS, tl2);
  }

  // ---- layer 0: gather-combine bases per edge + f32 atomics ----
  {
    size_t n16 = NH * sizeof(float) / 16;
    zero16_kernel<<<(unsigned)((n16 + 255) / 256), 256, 0, stream>>>((uint4*)bufA, n16);
  }
  l0_edge_kernel<<<eBlocks, edgeBlk, 0, stream>>>(src, dst, h, rel, norm, bases0, wcomp0, bufA, E, N);
  finish_kernel<<<nhBlocks, 256, 0, stream>>>(bufA, loop0, h, bias0, bufB, xbf, NH);

  // ---- layer 1: aggregate-first scatter (pk bf16 atomics), then WMMA GEMMs ----
  {
    size_t n16 = (size_t)RREL * NH * 2 / 16;
    zero16_kernel<<<(unsigned)((n16 + 255) / 256), 256, 0, stream>>>((uint4*)zbf, n16);
  }
  z_edge_kernel<<<eBlocks, edgeBlk, 0, stream>>>(src, dst, rel, norm, bufB, zbf, E, N);
  // out1 = x1 @ loop1 + bias1
  wmma_gemm_bf16<<<mBlocks, dim3(32, 8), 0, stream>>>(xbf, 0, 1, l1T, 0, nullptr, bias1, bufA, H128);
  // out1 += sum_r z[r] @ W1[r]
  wmma_gemm_bf16<<<mBlocks, dim3(32, 8), 0, stream>>>(zbf, NH, RREL, w1T, (size_t)H128 * H128,
                                                      bufA, nullptr, bufA, H128);
  finish_kernel<<<nhBlocks, 256, 0, stream>>>(bufA, nullptr, nullptr, nullptr, bufB, xbf, NH);

  // ---- layer 2 (H -> C, no relu) ----
  {
    size_t n16 = (size_t)RREL * NH * 2 / 16;
    zero16_kernel<<<(unsigned)((n16 + 255) / 256), 256, 0, stream>>>((uint4*)zbf, n16);
  }
  z_edge_kernel<<<eBlocks, edgeBlk, 0, stream>>>(src, dst, rel, norm, bufB, zbf, E, N);
  float* out = (float*)d_out;
  wmma_gemm_bf16<<<mBlocks, dim3(32, 1), 0, stream>>>(xbf, 0, 1, l2T, 0, nullptr, bias2, out, CCLS);
  wmma_gemm_bf16<<<mBlocks, dim3(32, 1), 0, stream>>>(zbf, NH, RREL, w2T, (size_t)CCLS * H128,
                                                      out, nullptr, out, CCLS);
  (void)out_size;
}
